// GNNLinkPredictor_45552423141686
// MI455X (gfx1250) — matrix operand order repaired and checked by reading, not compiled, not run on hardware
//
#include <hip/hip_runtime.h>
#include <hip/hip_bf16.h>

#define NN    100000
#define EMBD  128
#define HEADS 4
#define HIDC  32
#define C1    (HEADS * HIDC)   // 128
#define OUTC  64
#define NE    1600000
#define NL    100000
#define NSLOPE 0.2f

typedef __attribute__((ext_vector_type(16))) _Float16 v16h;
typedef __attribute__((ext_vector_type(8)))  _Float16 v8h;
typedef __attribute__((ext_vector_type(8)))  float    v8f;

// ---------- helpers ----------
__device__ __forceinline__ unsigned fmap(float x) {
    unsigned u = __float_as_uint(x);
    return (u & 0x80000000u) ? ~u : (u | 0x80000000u);
}
__device__ __forceinline__ float funmap(unsigned u) {
    return (u & 0x80000000u) ? __uint_as_float(u & 0x7FFFFFFFu) : __uint_as_float(~u);
}
__device__ __forceinline__ void fatomic_add(float* p, float v) {
    __hip_atomic_fetch_add(p, v, __ATOMIC_RELAXED, __HIP_MEMORY_SCOPE_AGENT);
}

// ---------- conversions ----------
__global__ void cvt_gather_f16(const float* __restrict__ emb, const int* __restrict__ x,
                               _Float16* __restrict__ out, int total, int C) {
    int t = blockIdx.x * blockDim.x + threadIdx.x;
    if (t >= total) return;
    int n = t / C, c = t - n * C;
    out[t] = (_Float16)emb[(size_t)x[n] * C + c];
}

// Pack row-major W[K,Nc] (f32) into WMMA B-fragment order (f16):
// Bp[((kt*ntiles + tn)*32 + lane)*16 + i]  where  K = kt*32 + (lane>>4)*16 + i,
// col = tn*16 + (lane&15).  Each wave-lane then reads its whole fragment with
// two contiguous b128 loads.
__global__ void pack_B_frag(const float* __restrict__ W, _Float16* __restrict__ Bp,
                            int K, int Nc) {
    int t = blockIdx.x * blockDim.x + threadIdx.x;
    if (t >= K * Nc) return;
    int i    = t & 15;
    int lane = (t >> 4) & 31;
    int tile = t >> 9;                 // kt*ntiles + tn
    int ntiles = Nc >> 4;
    int kt = tile / ntiles;
    int tn = tile - kt * ntiles;
    int k   = kt * 32 + (lane >> 4) * 16 + i;
    int col = tn * 16 + (lane & 15);
    Bp[t] = (_Float16)W[(size_t)k * Nc + col];
}

// ---------- WMMA GEMM: C[M,Nc] = A[M,K] (f16 row-major) * W (pre-packed fragments) ----
template <int K>
__global__ void wmma_gemm_f16(const _Float16* __restrict__ A, const _Float16* __restrict__ Bp,
                              float* __restrict__ C, int M, int Nc) {
    int wave = (blockIdx.x * blockDim.x + threadIdx.x) >> 5;
    int lane = threadIdx.x & 31;
    int ntile = Nc >> 4;
    int tm = wave / ntile;
    int tn = wave - tm * ntile;
    if (tm * 16 >= M) return;             // wave-uniform, EXEC stays full

    int r  = lane & 15;
    int hs = lane >> 4;                   // which half of the wave
    size_t rowA = (size_t)(tm * 16 + r) * K;

    v8f acc = {};
#pragma unroll
    for (int kt = 0; kt < K / 32; ++kt) {
        int k0 = kt * 32;
        // A fragment (16x32 f16): lanes 0-15 hold K [0..15], lanes 16-31 [16..31]
        v8h lo = *reinterpret_cast<const v8h*>(A + rowA + k0 + hs * 16);
        v8h hi = *reinterpret_cast<const v8h*>(A + rowA + k0 + hs * 16 + 8);
        v16h a;
#pragma unroll
        for (int i = 0; i < 8; ++i) { a[i] = lo[i]; a[8 + i] = hi[i]; }
        // B fragment: pre-packed, contiguous 16 halfs per lane
        const _Float16* bbase = Bp + ((size_t)(kt * ntile + tn) * 32 + lane) * 16;
        v8h b0 = *reinterpret_cast<const v8h*>(bbase);
        v8h b1 = *reinterpret_cast<const v8h*>(bbase + 8);
        v16h b;
#pragma unroll
        for (int i = 0; i < 8; ++i) { b[i] = b0[i]; b[8 + i] = b1[i]; }
        acc = __builtin_amdgcn_wmma_f32_16x16x32_f16(false, a, false, b, (short)0, acc, false, false);
    }
#pragma unroll
    for (int i = 0; i < 8; ++i) {
        int rowc = tm * 16 + i + 8 * hs;  // C layout: VGPR i -> M = i (lanes 0-15) / 8+i (16-31)
        C[(size_t)rowc * Nc + tn * 16 + r] = acc[i];
    }
}

// ---------- per-node attention logits ----------
__global__ void node_alpha(const float* __restrict__ Hm, const float* __restrict__ a_s,
                           const float* __restrict__ a_d, float* __restrict__ as_o,
                           float* __restrict__ ad_o, int n, int heads, int hid) {
    int t = blockIdx.x * blockDim.x + threadIdx.x;
    if (t >= n * heads) return;
    int node = t / heads, h = t - node * heads;
    const float* hp = Hm + (size_t)node * heads * hid + h * hid;
    float ss = 0.f, sd = 0.f;
    for (int c = 0; c < hid; ++c) {
        float v = hp[c];
        ss += v * a_s[h * hid + c];
        sd += v * a_d[h * hid + c];
    }
    as_o[t] = ss; ad_o[t] = sd;
}

// ---------- edge pass 1: segment max of leaky-relu scores over dst ----------
__global__ void edge_max_k(const int* __restrict__ ei, int E, int n_nodes,
                           const float* __restrict__ as_i, const float* __restrict__ ad_i,
                           unsigned* __restrict__ m, int heads) {
    int t = blockIdx.x * blockDim.x + threadIdx.x;
    int total = (E + n_nodes) * heads;
    if (t >= total) return;
    int eg = t / heads, h = t - eg * heads;
    int s, d;
    if (eg < E) { s = ei[eg]; d = ei[E + eg]; } else { s = d = eg - E; }
    float sc = as_i[(size_t)s * heads + h] + ad_i[(size_t)d * heads + h];
    sc = sc > 0.f ? sc : NSLOPE * sc;
    atomicMax(&m[(size_t)d * heads + h], fmap(sc));
}

// ---------- edge pass 2: segment sum of exp(score - max) ----------
__global__ void edge_sum_k(const int* __restrict__ ei, int E, int n_nodes,
                           const float* __restrict__ as_i, const float* __restrict__ ad_i,
                           const unsigned* __restrict__ m, float* __restrict__ den, int heads) {
    int t = blockIdx.x * blockDim.x + threadIdx.x;
    int total = (E + n_nodes) * heads;
    if (t >= total) return;
    int eg = t / heads, h = t - eg * heads;
    int s, d;
    if (eg < E) { s = ei[eg]; d = ei[E + eg]; } else { s = d = eg - E; }
    float sc = as_i[(size_t)s * heads + h] + ad_i[(size_t)d * heads + h];
    sc = sc > 0.f ? sc : NSLOPE * sc;
    float ex = expf(sc - funmap(m[(size_t)d * heads + h]));
    fatomic_add(&den[(size_t)d * heads + h], ex);
}

// ---------- edge pass 3: weighted scatter-add aggregation (one wave per edge) ----------
// Lanes 0..heads-1 compute the per-head alpha once; it is broadcast with __shfl.
__global__ void edge_agg_k(const int* __restrict__ ei, int E, int n_nodes,
                           const float* __restrict__ as_i, const float* __restrict__ ad_i,
                           const unsigned* __restrict__ m, const float* __restrict__ den,
                           const float* __restrict__ Hm, float* __restrict__ agg,
                           int heads, int Ctot) {
    int wave = (blockIdx.x * blockDim.x + threadIdx.x) >> 5;
    int lane = threadIdx.x & 31;
    int Eall = E + n_nodes;
    if (wave >= Eall) return;
    int s, d;
    if (wave < E) { s = ei[wave]; d = ei[E + wave]; } else { s = d = wave - E; }

    float alpha_l = 0.f;
    if (lane < heads) {
        float sc = as_i[(size_t)s * heads + lane] + ad_i[(size_t)d * heads + lane];
        sc = sc > 0.f ? sc : NSLOPE * sc;
        float mv = funmap(m[(size_t)d * heads + lane]);
        alpha_l = expf(sc - mv) / (den[(size_t)d * heads + lane] + 1e-16f);
    }
    int hid = Ctot / heads;
    for (int c = lane; c < Ctot; c += 32) {
        float alpha = __shfl(alpha_l, c / hid, 32);
        fatomic_add(&agg[(size_t)d * Ctot + c], Hm[(size_t)s * Ctot + c] * alpha);
    }
}

// ---------- bias + ELU + f16 convert ----------
__global__ void elu_cvt(const float* __restrict__ agg, const float* __restrict__ bias,
                        _Float16* __restrict__ out, int total, int C) {
    int t = blockIdx.x * blockDim.x + threadIdx.x;
    if (t >= total) return;
    float v = agg[t] + bias[t % C];
    v = v > 0.f ? v : (expf(v) - 1.f);
    out[t] = (_Float16)v;
}

// ---------- decode: dot-product link scores (one wave per label edge) ----------
__global__ void decode_k(const int* __restrict__ eli, int nl, const float* __restrict__ Z,
                         const float* __restrict__ b2, float* __restrict__ out, int C) {
    int wave = (blockIdx.x * blockDim.x + threadIdx.x) >> 5;
    int lane = threadIdx.x & 31;
    if (wave >= nl) return;
    int a = eli[wave], b = eli[nl + wave];
    float p = 0.f;
    for (int c = lane; c < C; c += 32) {
        p += (Z[(size_t)a * C + c] + b2[c]) * (Z[(size_t)b * C + c] + b2[c]);
    }
#pragma unroll
    for (int off = 16; off > 0; off >>= 1) p += __shfl_xor(p, off, 32);
    if (lane == 0) out[wave] = p;
}

extern "C" void kernel_launch(void* const* d_in, const int* in_sizes, int n_in,
                              void* d_out, int out_size, void* d_ws, size_t ws_size,
                              hipStream_t stream) {
    (void)in_sizes; (void)n_in; (void)out_size; (void)ws_size;

    const int*   x    = (const int*)d_in[0];
    const int*   ei   = (const int*)d_in[1];
    const int*   eli  = (const int*)d_in[2];
    const float* emb  = (const float*)d_in[3];
    const float* W1   = (const float*)d_in[4];
    const float* as1w = (const float*)d_in[5];
    const float* ad1w = (const float*)d_in[6];
    const float* b1   = (const float*)d_in[7];
    const float* W2   = (const float*)d_in[8];
    const float* as2w = (const float*)d_in[9];
    const float* ad2w = (const float*)d_in[10];
    const float* b2   = (const float*)d_in[11];
    float* out = (float*)d_out;

    // ---- workspace carve-out (with reuse) ----
    char* p = (char*)d_ws;
    auto carve = [&](size_t bytes) { char* r = p; p += (bytes + 255) & ~(size_t)255; return (void*)r; };
    _Float16* emb_h = (_Float16*)carve((size_t)NN * EMBD * 2); // reused as h1_elu_f16 after GEMM1
    _Float16* W1p   = (_Float16*)carve((size_t)EMBD * C1 * 2); // packed B fragments
    _Float16* W2p   = (_Float16*)carve((size_t)C1 * OUTC * 2); // packed B fragments
    float*    H1    = (float*)carve((size_t)NN * C1 * 4);      // reused as H2 (NN x 64)
    float*    AGG1  = (float*)carve((size_t)NN * C1 * 4);      // reused as Z  (NN x 64)
    float*    as1   = (float*)carve((size_t)NN * HEADS * 4);
    float*    ad1   = (float*)carve((size_t)NN * HEADS * 4);
    unsigned* m1    = (unsigned*)carve((size_t)NN * HEADS * 4);
    float*    dn1   = (float*)carve((size_t)NN * HEADS * 4);
    float*    as2   = (float*)carve((size_t)NN * 4);
    float*    ad2   = (float*)carve((size_t)NN * 4);
    unsigned* m2    = (unsigned*)carve((size_t)NN * 4);
    float*    dn2   = (float*)carve((size_t)NN * 4);

    const int B = 256;
    const int Eall = NE + NN;

    // ---- f16 conversions: gathered embeddings + pre-swizzled weight fragments ----
    {
        int tot = NN * EMBD;
        cvt_gather_f16<<<(tot + B - 1) / B, B, 0, stream>>>(emb, x, emb_h, tot, EMBD);
        pack_B_frag<<<(EMBD * C1 + B - 1) / B, B, 0, stream>>>(W1, W1p, EMBD, C1);
        pack_B_frag<<<(C1 * OUTC + B - 1) / B, B, 0, stream>>>(W2, W2p, C1, OUTC);
    }

    // ---- layer 1: H1 = emb @ W1 (WMMA, K=128 fully unrolled) ----
    {
        int waves = (NN / 16) * (C1 / 16);              // 50000 waves
        wmma_gemm_f16<EMBD><<<(waves * 32 + B - 1) / B, B, 0, stream>>>(emb_h, W1p, H1, NN, C1);
        node_alpha<<<(NN * HEADS + B - 1) / B, B, 0, stream>>>(H1, as1w, ad1w, as1, ad1, NN, HEADS, HIDC);
        hipMemsetAsync(m1, 0, (size_t)NN * HEADS * 4, stream);
        hipMemsetAsync(dn1, 0, (size_t)NN * HEADS * 4, stream);
        hipMemsetAsync(AGG1, 0, (size_t)NN * C1 * 4, stream);
        int etot = Eall * HEADS;
        edge_max_k<<<(etot + B - 1) / B, B, 0, stream>>>(ei, NE, NN, as1, ad1, m1, HEADS);
        edge_sum_k<<<(etot + B - 1) / B, B, 0, stream>>>(ei, NE, NN, as1, ad1, m1, dn1, HEADS);
        edge_agg_k<<<(Eall * 32 + B - 1) / B, B, 0, stream>>>(ei, NE, NN, as1, ad1, m1, dn1,
                                                             H1, AGG1, HEADS, C1);
        // bias + ELU + convert into emb_h (reused as f16 activations)
        int tot = NN * C1;
        elu_cvt<<<(tot + B - 1) / B, B, 0, stream>>>(AGG1, b1, emb_h, tot, C1);
    }

    // ---- layer 2: H2 = h1_elu @ W2 (WMMA), Z = attention-aggregate ----
    float* H2 = H1;    // NN x 64, reuses H1 region
    float* Z  = AGG1;  // NN x 64, reuses AGG1 region
    {
        hipMemsetAsync(Z, 0, (size_t)NN * OUTC * 4, stream);
        hipMemsetAsync(m2, 0, (size_t)NN * 4, stream);
        hipMemsetAsync(dn2, 0, (size_t)NN * 4, stream);
        int waves = (NN / 16) * (OUTC / 16);            // 25000 waves
        wmma_gemm_f16<C1><<<(waves * 32 + B - 1) / B, B, 0, stream>>>(emb_h, W2p, H2, NN, OUTC);
        node_alpha<<<(NN + B - 1) / B, B, 0, stream>>>(H2, as2w, ad2w, as2, ad2, NN, 1, OUTC);
        edge_max_k<<<(Eall + B - 1) / B, B, 0, stream>>>(ei, NE, NN, as2, ad2, m2, 1);
        edge_sum_k<<<(Eall + B - 1) / B, B, 0, stream>>>(ei, NE, NN, as2, ad2, m2, dn2, 1);
        edge_agg_k<<<(Eall * 32 + B - 1) / B, B, 0, stream>>>(ei, NE, NN, as2, ad2, m2, dn2,
                                                             H2, Z, 1, OUTC);
    }

    // ---- decode: link scores ----
    decode_k<<<(NL * 32 + B - 1) / B, B, 0, stream>>>(eli, NL, Z, b2, out, OUTC);
}
